// GNN_2276332667421
// MI455X (gfx1250) — matrix operand order repaired and checked by reading, not compiled, not run on hardware
//
#include <hip/hip_runtime.h>
#include <hip/hip_bf16.h>

#define N_NODES 50000
#define M_EDGES 800000
#define DD      128
#define MSGW    128
#define EFW     32
#define IN_DIM  290
#define K_PAD   320   // IN_DIM padded to multiple of 32
#define EPB     32    // edges per block (2 row tiles of 16)

typedef _Float16 h8  __attribute__((ext_vector_type(8)));
typedef _Float16 h16 __attribute__((ext_vector_type(16)));
typedef float    f8  __attribute__((ext_vector_type(8)));

// ---- WMMA fragment loaders (wave32, 16x16x32 f16) ------------------------
// A (16xK, row per lane): lane m=lane&15 holds row m; K = kbase +
// half*16 + e (contiguous 16 f16, half=lane>>4).
__device__ __forceinline__ h16 loadA(const _Float16* row, int half, int kbase) {
  return *(const h16*)(row + kbase + half * 16);
}
// B (Kx16, column per lane): lane n=lane&15 holds column n; K = kbase +
// half*16 + e (contiguous 16 f16). `col` points at the column's K=0.
__device__ __forceinline__ h16 loadB(const _Float16* col, int half, int kbase) {
  return *(const h16*)(col + kbase + half * 16);
}

__device__ __forceinline__ f8 wmma16(h16 a, h16 b, f8 c) {
  return __builtin_amdgcn_wmma_f32_16x16x32_f16(false, a, false, b,
                                                (short)0, c, false, false);
}

__device__ __forceinline__ float sigm(float x) { return 1.0f / (1.0f + __expf(-x)); }

// ---- prep kernels --------------------------------------------------------
__global__ __launch_bounds__(256) void k_zero(float* p, int n) {
  int i = blockIdx.x * 256 + threadIdx.x;
  if (i < n) p[i] = 0.f;
}
// w: (IN_DIM,128) row-major -> o: (128, K_PAD) f16, o[oc*K_PAD+k] = w[k*128+oc]
__global__ __launch_bounds__(256) void k_pack_w1(const float* __restrict__ w,
                                                 _Float16* __restrict__ o) {
  int i = blockIdx.x * 256 + threadIdx.x;
  if (i >= 128 * K_PAD) return;
  int oc = i / K_PAD, k = i - oc * K_PAD;
  o[i] = (_Float16)((k < IN_DIM) ? w[k * 128 + oc] : 0.f);
}
// w: (128,128) -> transposed f16
__global__ __launch_bounds__(256) void k_pack_w2(const float* __restrict__ w,
                                                 _Float16* __restrict__ o) {
  int i = blockIdx.x * 256 + threadIdx.x;
  if (i >= 128 * 128) return;
  int oc = i >> 7, k = i & 127;
  o[i] = (_Float16)w[k * 128 + oc];
}
__global__ __launch_bounds__(256) void k_cvt16(const float* __restrict__ w,
                                               _Float16* __restrict__ o, int n) {
  int i = blockIdx.x * 256 + threadIdx.x;
  if (i < n) o[i] = (_Float16)w[i];
}

// ---- edge message kernel: 32 edges / block, 8 waves ----------------------
// Each wave owns one 16-wide output column tile and computes TWO 16-edge row
// tiles per loaded B fragment (2x B reuse -> 4 WMMAs per kc in each layer).
__global__ __launch_bounds__(256) void k_edges(
    const float* __restrict__ state, const int* __restrict__ edge,
    const float* __restrict__ efeat, const float* __restrict__ nattr,
    const float* __restrict__ eattr,
    const _Float16* __restrict__ w1m, const _Float16* __restrict__ w1a,
    const _Float16* __restrict__ w2m, const _Float16* __restrict__ w2a,
    const float* __restrict__ b1m, const float* __restrict__ b1a,
    const float* __restrict__ b2m, const float* __restrict__ b2a,
    float* __restrict__ agg) {
  __shared__ __align__(32) _Float16 sIn[EPB * K_PAD];
  __shared__ __align__(32) _Float16 sH[EPB * MSGW];
  __shared__ __align__(32) _Float16 sA[EPB * MSGW];
  __shared__ int sSrc[EPB], sDst[EPB];

  const int tid = threadIdx.x;
  const int tileE = blockIdx.x * EPB;

  if (tid < EPB) {
    int e = tileE + tid, s = 0, d = 0;
    if (e < M_EDGES) { s = edge[2 * e]; d = edge[2 * e + 1]; }
    sSrc[tid] = s; sDst[tid] = d;
  }
  __syncthreads();

  // Stage edge_input tile (f16, zero-padded to K_PAD)
  for (int idx = tid; idx < EPB * K_PAD; idx += 256) {
    int r = idx / K_PAD, c = idx - r * K_PAD;
    int e = tileE + r;
    float v = 0.f;
    if (e < M_EDGES) {
      int s = sSrc[r], d = sDst[r];
      if (c < DD)                 v = state[s * DD + c] - state[d * DD + c];
      else if (c < DD + EFW)      v = efeat[e * EFW + (c - DD)];
      else if (c < 2 * DD + EFW)  { int k = c - (DD + EFW);
                                    v = eattr[s * DD + k] - eattr[d * DD + k]; }
      else if (c < IN_DIM)        { int k = c - (2 * DD + EFW);
                                    v = nattr[k * N_NODES + s] - nattr[k * N_NODES + d]; }
    }
    sIn[idx] = (_Float16)v;
  }
  __syncthreads();

  const int wave = tid >> 5, lane = tid & 31;
  const int half = lane >> 4, ln = lane & 15;
  const int colg = wave * 16 + ln;  // output column 0..127

  // Layer 1: h = relu(x@W1m + b), a = relu(x@W1a + b); 2 row tiles per wave
  f8 ch0 = {}, ch1 = {}, ca0 = {}, ca1 = {};
#pragma unroll
  for (int kc = 0; kc < K_PAD / 32; ++kc) {
    h16 bm = loadB(w1m + colg * K_PAD, half, kc * 32);
    h16 ba = loadB(w1a + colg * K_PAD, half, kc * 32);
    h16 a0 = loadA(sIn + ln * K_PAD, half, kc * 32);
    h16 a1 = loadA(sIn + (16 + ln) * K_PAD, half, kc * 32);
    ch0 = wmma16(a0, bm, ch0);
    ch1 = wmma16(a1, bm, ch1);
    ca0 = wmma16(a0, ba, ca0);
    ca1 = wmma16(a1, ba, ca1);
  }
  {
    float bh = b1m[colg], bv = b1a[colg];
#pragma unroll
    for (int r = 0; r < 8; ++r) {
      int row = r + half * 8;
      float x0 = ch0[r] + bh; x0 = x0 > 0.f ? x0 : 0.f;
      float x1 = ch1[r] + bh; x1 = x1 > 0.f ? x1 : 0.f;
      float y0 = ca0[r] + bv; y0 = y0 > 0.f ? y0 : 0.f;
      float y1 = ca1[r] + bv; y1 = y1 > 0.f ? y1 : 0.f;
      sH[row * MSGW + colg] = (_Float16)x0;
      sH[(16 + row) * MSGW + colg] = (_Float16)x1;
      sA[row * MSGW + colg] = (_Float16)y0;
      sA[(16 + row) * MSGW + colg] = (_Float16)y1;
    }
  }
  __syncthreads();

  // Layer 2: msg = h@W2m + b ; att = sigmoid(a@W2a + b) ; scatter msg*att
  f8 cm0 = {}, cm1 = {}, cs0 = {}, cs1 = {};
#pragma unroll
  for (int kc = 0; kc < MSGW / 32; ++kc) {
    h16 bm = loadB(w2m + colg * MSGW, half, kc * 32);
    h16 bs = loadB(w2a + colg * MSGW, half, kc * 32);
    h16 am0 = loadA(sH + ln * MSGW, half, kc * 32);
    h16 am1 = loadA(sH + (16 + ln) * MSGW, half, kc * 32);
    h16 aa0 = loadA(sA + ln * MSGW, half, kc * 32);
    h16 aa1 = loadA(sA + (16 + ln) * MSGW, half, kc * 32);
    cm0 = wmma16(am0, bm, cm0);
    cm1 = wmma16(am1, bm, cm1);
    cs0 = wmma16(aa0, bs, cs0);
    cs1 = wmma16(aa1, bs, cs1);
  }
  {
    float bm2 = b2m[colg], bs2 = b2a[colg];
#pragma unroll
    for (int r = 0; r < 8; ++r) {
      int row = r + half * 8;
      int e0 = tileE + row;
      int e1 = tileE + 16 + row;
      if (e0 < M_EDGES) {
        float m = cm0[r] + bm2;
        float t = sigm(cs0[r] + bs2);
        atomicAdd(&agg[(size_t)sDst[row] * MSGW + colg], m * t);
      }
      if (e1 < M_EDGES) {
        float m = cm1[r] + bm2;
        float t = sigm(cs1[r] + bs2);
        atomicAdd(&agg[(size_t)sDst[16 + row] * MSGW + colg], m * t);
      }
    }
  }
}

// ---- node GRU kernel: 16 nodes / block, 8 waves --------------------------
__global__ __launch_bounds__(256) void k_nodes(
    const float* __restrict__ state, const float* __restrict__ agg,
    const _Float16* __restrict__ wih, const _Float16* __restrict__ whh,
    const float* __restrict__ bih, const float* __restrict__ bhh,
    float* __restrict__ out) {
  __shared__ __align__(32) _Float16 sAgg[16 * DD];
  __shared__ __align__(32) _Float16 sSt[16 * DD];
  __shared__ float sGi[16 * 3 * DD];
  __shared__ float sGh[16 * 3 * DD];

  const int tid = threadIdx.x;
  const int tileN = blockIdx.x * 16;

  for (int idx = tid; idx < 16 * DD; idx += 256) {
    int r = idx >> 7, c = idx & 127;
    int g = tileN + r;
    float va = 0.f, vs = 0.f;
    if (g < N_NODES) { va = agg[(size_t)g * DD + c]; vs = state[(size_t)g * DD + c]; }
    sAgg[idx] = (_Float16)va;
    sSt[idx] = (_Float16)vs;
  }
  __syncthreads();

  const int wave = tid >> 5, lane = tid & 31;
  const int half = lane >> 4, ln = lane & 15;

#pragma unroll
  for (int tt = 0; tt < 3; ++tt) {
    int t = wave + tt * 8;          // 0..23 of the 384 output columns
    int colg = t * 16 + ln;
    f8 ci = {}; f8 chh = {};
#pragma unroll
    for (int kc = 0; kc < DD / 32; ++kc) {
      h16 aI = loadA(sAgg + ln * DD, half, kc * 32);
      h16 aH = loadA(sSt + ln * DD, half, kc * 32);
      h16 bI = loadB(wih + colg * DD, half, kc * 32);
      h16 bH = loadB(whh + colg * DD, half, kc * 32);
      ci = wmma16(aI, bI, ci);
      chh = wmma16(aH, bH, chh);
    }
#pragma unroll
    for (int r = 0; r < 8; ++r) {
      int row = r + half * 8;
      sGi[row * 384 + colg] = ci[r];
      sGh[row * 384 + colg] = chh[r];
    }
  }
  __syncthreads();

  for (int idx = tid; idx < 16 * DD; idx += 256) {
    int r = idx >> 7, c = idx & 127;
    int g = tileN + r;
    if (g >= N_NODES) continue;
    float ir = sGi[r * 384 + c] + bih[c];
    float iz = sGi[r * 384 + 128 + c] + bih[128 + c];
    float in_ = sGi[r * 384 + 256 + c] + bih[256 + c];
    float hr = sGh[r * 384 + c] + bhh[c];
    float hz = sGh[r * 384 + 128 + c] + bhh[128 + c];
    float hn = sGh[r * 384 + 256 + c] + bhh[256 + c];
    float rg = sigm(ir + hr);
    float zg = sigm(iz + hz);
    float ng = tanhf(in_ + rg * hn);
    float st = state[(size_t)g * DD + c];
    out[(size_t)g * DD + c] = (1.f - zg) * ng + zg * st;
  }
}

// ---- launcher ------------------------------------------------------------
extern "C" void kernel_launch(void* const* d_in, const int* in_sizes, int n_in,
                              void* d_out, int out_size, void* d_ws, size_t ws_size,
                              hipStream_t stream) {
  const float* state  = (const float*)d_in[0];
  const int*   edge   = (const int*)d_in[1];
  const float* efeat  = (const float*)d_in[2];
  const float* nattr  = (const float*)d_in[3];
  const float* eattr  = (const float*)d_in[4];
  const float* msg_w1 = (const float*)d_in[5];
  const float* msg_b1 = (const float*)d_in[6];
  const float* msg_w2 = (const float*)d_in[7];
  const float* msg_b2 = (const float*)d_in[8];
  const float* att_w1 = (const float*)d_in[9];
  const float* att_b1 = (const float*)d_in[10];
  const float* att_w2 = (const float*)d_in[11];
  const float* att_b2 = (const float*)d_in[12];
  const float* gwih   = (const float*)d_in[13];
  const float* gwhh   = (const float*)d_in[14];
  const float* gbih   = (const float*)d_in[15];
  const float* gbhh   = (const float*)d_in[16];

  char* wsb = (char*)d_ws;
  size_t off = 0;
  auto alloc = [&](size_t bytes) {
    void* p = wsb + off;
    off = (off + bytes + 255) & ~(size_t)255;
    return p;
  };
  float*    agg = (float*)alloc((size_t)N_NODES * MSGW * sizeof(float));
  _Float16* w1m = (_Float16*)alloc((size_t)128 * K_PAD * 2);
  _Float16* w1a = (_Float16*)alloc((size_t)128 * K_PAD * 2);
  _Float16* w2m = (_Float16*)alloc((size_t)128 * 128 * 2);
  _Float16* w2a = (_Float16*)alloc((size_t)128 * 128 * 2);
  _Float16* wih = (_Float16*)alloc((size_t)384 * 128 * 2);
  _Float16* whh = (_Float16*)alloc((size_t)384 * 128 * 2);

  // Prep: zero accumulator, convert/transpose weights to f16 B-layout.
  k_zero<<<(N_NODES * MSGW + 255) / 256, 256, 0, stream>>>(agg, N_NODES * MSGW);
  k_pack_w1<<<(128 * K_PAD + 255) / 256, 256, 0, stream>>>(msg_w1, w1m);
  k_pack_w1<<<(128 * K_PAD + 255) / 256, 256, 0, stream>>>(att_w1, w1a);
  k_pack_w2<<<(128 * 128 + 255) / 256, 256, 0, stream>>>(msg_w2, w2m);
  k_pack_w2<<<(128 * 128 + 255) / 256, 256, 0, stream>>>(att_w2, w2a);
  k_cvt16<<<(384 * 128 + 255) / 256, 256, 0, stream>>>(gwih, wih, 384 * 128);
  k_cvt16<<<(384 * 128 + 255) / 256, 256, 0, stream>>>(gwhh, whh, 384 * 128);

  // Edge MLP + attention + scatter-add
  k_edges<<<(M_EDGES + EPB - 1) / EPB, 256, 0, stream>>>(
      state, edge, efeat, nattr, eattr, w1m, w1a, w2m, w2a,
      msg_b1, att_b1, msg_b2, att_b2, agg);

  // GRU node update
  k_nodes<<<(N_NODES + 15) / 16, 256, 0, stream>>>(
      state, agg, wih, whh, gbih, gbhh, (float*)d_out);
}
